// Attention_69758858821855
// MI455X (gfx1250) — matrix operand order repaired and checked by reading, not compiled, not run on hardware
//
#include <hip/hip_runtime.h>

typedef __attribute__((ext_vector_type(16))) _Float16 v16h;
typedef __attribute__((ext_vector_type(8)))  _Float16 v8h;
typedef __attribute__((ext_vector_type(8)))  float    v8f;
typedef __attribute__((ext_vector_type(4)))  float    v4f;
typedef __attribute__((ext_vector_type(4)))  unsigned int v4u;
typedef __attribute__((ext_vector_type(8)))  int      v8i;
typedef __attribute__((ext_vector_type(4)))  int      v4i;

#define S_LEN  4096
#define DMODEL 512
#define NHEAD  8
#define DHEAD  64

#if __has_builtin(__builtin_amdgcn_tensor_load_to_lds)
#define CDNA5_TDM 1
#else
#define CDNA5_TDM 0
#endif

__device__ __forceinline__ v8f wmma_f16(v16h a, v16h b, v8f c) {
  return __builtin_amdgcn_wmma_f32_16x16x32_f16(false, a, false, b, (short)0, c,
                                                false, false);
}

__device__ __forceinline__ v16h pack16(v8h lo, v8h hi) {
  v16h r;
#pragma unroll
  for (int j = 0; j < 8; ++j) { r[j] = lo[j]; r[8 + j] = hi[j]; }
  return r;
}

#if CDNA5_TDM
// Issue one TDM 2D tile load (global -> LDS).  D# per ISA ch.8:
// group0 = {count=1, lds_addr, global_addr, type=2}, group1 = dims/strides.
// data_size = 2 bytes (f16).  Tracked on TENSORcnt.
__device__ __forceinline__ void tdm_load_2d(unsigned ldsOff, const void* gptr,
                                            unsigned tensorDim0, unsigned tensorDim1,
                                            unsigned tileDim0, unsigned tileDim1,
                                            unsigned stride0) {
  unsigned long long ga = (unsigned long long)gptr;
  v4u g0;
  g0[0] = 1u;                                    // count=1, user descriptor
  g0[1] = ldsOff;                                // LDS byte address
  g0[2] = (unsigned)ga;                          // global_addr[31:0]
  g0[3] = ((unsigned)(ga >> 32) & 0x01FFFFFFu) | (2u << 30);  // addr[56:32] | type=2
  v8i g1;
  g1[0] = (int)(1u << 16);                       // data_size=1 -> 2 bytes
  g1[1] = (int)(tensorDim0 << 16);               // tensor_dim0 [79:48] lo
  g1[2] = (int)((tensorDim0 >> 16) | (tensorDim1 << 16));
  g1[3] = (int)((tensorDim1 >> 16) | (tileDim0 << 16));   // tile_dim0 [127:112]
  g1[4] = (int)(tileDim1 & 0xFFFFu);             // tile_dim1 [143:128], tile_dim2=0
  g1[5] = (int)stride0;                          // tensor_dim0_stride [207:160]
  g1[6] = 0;
  g1[7] = 0;
  v4i z4 = {0, 0, 0, 0};
#if __clang_major__ >= 23
  v8i z8 = {0, 0, 0, 0, 0, 0, 0, 0};
  __builtin_amdgcn_tensor_load_to_lds(g0, g1, z4, z4, z8, 0);
#else
  __builtin_amdgcn_tensor_load_to_lds(g0, g1, z4, z4, 0);
#endif
}
#endif

// ---------------------------------------------------------------------------
// Kernel 0: f32 -> f16 conversion (vectorized, 8 elems/thread)
// ---------------------------------------------------------------------------
__global__ void cvt_kernel(const float* __restrict__ src,
                           _Float16* __restrict__ dst, int n8) {
  const int i = blockIdx.x * blockDim.x + threadIdx.x;
  if (i >= n8) return;
  v4f a = ((const v4f*)src)[2 * i];
  v4f b = ((const v4f*)src)[2 * i + 1];
  v8h o;
#pragma unroll
  for (int j = 0; j < 4; ++j) {
    o[j]     = (_Float16)a[j];
    o[4 + j] = (_Float16)b[j];
  }
  ((v8h*)dst)[i] = o;
}

// ---------------------------------------------------------------------------
// Kernel 1: fused QKV projection, f16 in, 16x64 tile per wave (4 WMMA / A-load)
// Q,K -> [B,H,S,DH] f16 ; V -> transposed [B,H,DH,S] f16.
// ---------------------------------------------------------------------------
__global__ void __launch_bounds__(256)
qkv_kernel(const _Float16* __restrict__ xh,
           const _Float16* __restrict__ wq,
           const _Float16* __restrict__ wk,
           const _Float16* __restrict__ wv,
           _Float16* __restrict__ qb,
           _Float16* __restrict__ kb,
           _Float16* __restrict__ vt) {
  const int lane = threadIdx.x & 31;
  const int wid  = blockIdx.x * (blockDim.x >> 5) + (threadIdx.x >> 5);
  if (wid >= 3 * 512 * 8) return;
  const int mat     = wid >> 12;       // 0=Q 1=K 2=V (4096 tiles each)
  const int rem     = wid & 4095;
  const int rowTile = rem >> 3;        // 0..511 token tiles
  const int cg      = rem & 7;         // 64-wide col group == head index
  const int half = lane >> 4;
  const int lq   = lane & 15;

  const _Float16* W  = (mat == 0) ? wq : (mat == 1) ? wk : wv;
  const _Float16* ar = xh + (size_t)(rowTile * 16 + lq) * DMODEL;
  const _Float16* wr = W + (size_t)(cg * 64 + lq) * DMODEL;   // + t*16*DMODEL

  v8f c[4] = {};
#pragma unroll 4
  for (int it = 0; it < 16; ++it) {
    const int ka = it * 32 + 16 * half;
    v16h a = pack16(*(const v8h*)(ar + ka), *(const v8h*)(ar + ka + 8));
    const int kbb = it * 32 + 16 * half;
#pragma unroll
    for (int t = 0; t < 4; ++t) {
      const _Float16* w = wr + (size_t)t * 16 * DMODEL + kbb;
      v16h b = pack16(*(const v8h*)(w), *(const v8h*)(w + 8));
      c[t] = wmma_f16(a, b, c[t]);
    }
  }

  const int sTile = rowTile * 16;
  const int bIdx  = sTile >> 12;
  const int sLoc  = (sTile & (S_LEN - 1)) + 8 * half;
  const int h     = cg;
  if (mat < 2) {
    _Float16* base = (mat == 0 ? qb : kb) +
                     ((size_t)(bIdx * NHEAD + h) * S_LEN + sLoc) * DHEAD;
#pragma unroll
    for (int t = 0; t < 4; ++t) {
      const int dh = t * 16 + lq;
#pragma unroll
      for (int r = 0; r < 8; ++r) base[(size_t)r * DHEAD + dh] = (_Float16)c[t][r];
    }
  } else {
#pragma unroll
    for (int t = 0; t < 4; ++t) {
      const int dh = t * 16 + lq;
      _Float16* dst = vt + ((size_t)(bIdx * NHEAD + h) * DHEAD + dh) * S_LEN + sLoc;
      v8h pk;
#pragma unroll
      for (int r = 0; r < 8; ++r) pk[r] = (_Float16)c[t][r];
      *(v8h*)dst = pk;
    }
  }
}

// ---------------------------------------------------------------------------
// Kernel 2: causal flash attention. Block = 8 waves = 128 queries of one (b,h).
// K/V 32-key chunks staged into LDS once per block via the Tensor Data Mover;
// S^T = K @ Q^T so softmaxed P lands directly in the A-operand layout of P@V.
// ---------------------------------------------------------------------------
__global__ void __launch_bounds__(256)
flash_kernel(const _Float16* __restrict__ Q,
             const _Float16* __restrict__ K,
             const _Float16* __restrict__ Vt,
             _Float16* __restrict__ attn) {
  const int lane = threadIdx.x & 31;
  const int wave = threadIdx.x >> 5;
  const int bh   = blockIdx.x >> 5;         // 16 (b,h) slices
  const int qblk = blockIdx.x & 31;         // 32 blocks x 128 queries
  const int half = lane >> 4;
  const int lq   = lane & 15;

  const _Float16* Qp = Q  + (size_t)bh * S_LEN * DHEAD;
  const _Float16* Kp = K  + (size_t)bh * S_LEN * DHEAD;
  const _Float16* Vp = Vt + (size_t)bh * DHEAD * S_LEN;
  const int qBaseBlk = qblk * 128;
  const int qBase    = qBaseBlk + wave * 16;

#if CDNA5_TDM
  extern __shared__ _Float16 smem[];
  const _Float16* smemK = smem;            // 32 keys x 64 dh  (4 KB)
  const _Float16* smemV = smem + 2048;     // 64 dh  x 32 keys (4 KB)
#endif

  // Q^T as B operand (k-dim = dh): b[j] = Q[qBase+lq][chunk*32 + 16*(j>>3) + 8*half + (j&7)]
  const _Float16* qr = Qp + (size_t)(qBase + lq) * DHEAD + 8 * half;
  v16h bq0 = pack16(*(const v8h*)(qr),      *(const v8h*)(qr + 16));
  v16h bq1 = pack16(*(const v8h*)(qr + 32), *(const v8h*)(qr + 48));

  v8f o[4] = {};
  float m = -3.0e38f, lsum = 0.f;
  const int q       = qBase + lq;
  const int myKcMax = (qBase + 15) >> 5;
  const int kcMax   = (qBaseBlk + 127) >> 5;   // uniform across block

  for (int kc = 0; kc <= kcMax; ++kc) {
    const int kBase = kc * 32;
#if CDNA5_TDM
    __syncthreads();                            // prior-iter LDS reads done
    if (wave == 0) {
      // K chunk: tensor [S x 64], tile 64x32 @ row kBase -> LDS rows stride 64
      tdm_load_2d(0, Kp + (size_t)kBase * DHEAD, DHEAD, S_LEN, DHEAD, 32, DHEAD);
      // V chunk: tensor [64 x S], tile 32x64 @ col kBase -> LDS rows stride 32
      tdm_load_2d(4096, Vp + kBase, S_LEN, DHEAD, 32, DHEAD, S_LEN);
      __builtin_amdgcn_s_wait_tensorcnt(0);
    }
    __syncthreads();                            // LDS tiles visible to all waves
    const _Float16* kTile = smemK;              // row stride 64
    const _Float16* vTile = smemV;              // row stride 32
    const int vstr = 32;
#else
    const _Float16* kTile = Kp + (size_t)kBase * DHEAD;
    const _Float16* vTile = Vp + kBase;
    const int vstr = S_LEN;
#endif
    if (kc <= myKcMax) {
      // K as A operand: a[j] -> dh = dhc*32 + 16*(j>>3) + 8*half + (j&7)
      const _Float16* kr0 = kTile + (size_t)lq * DHEAD + 8 * half;
      const _Float16* kr1 = kr0 + 16 * DHEAD;
      v16h a00 = pack16(*(const v8h*)(kr0),      *(const v8h*)(kr0 + 16));
      v16h a01 = pack16(*(const v8h*)(kr0 + 32), *(const v8h*)(kr0 + 48));
      v16h a10 = pack16(*(const v8h*)(kr1),      *(const v8h*)(kr1 + 16));
      v16h a11 = pack16(*(const v8h*)(kr1 + 32), *(const v8h*)(kr1 + 48));

      v8f st0 = {}, st1 = {};
      st0 = wmma_f16(a00, bq0, st0);
      st0 = wmma_f16(a01, bq1, st0);            // keys kBase..+15
      st1 = wmma_f16(a10, bq0, st1);
      st1 = wmma_f16(a11, bq1, st1);            // keys kBase+16..+31

      float cmax = -3.0e38f;
#pragma unroll
      for (int r = 0; r < 8; ++r) {
        const int key0 = kBase + r + 8 * half;
        float s0 = (float)st0[r] * 0.125f;
        if (key0 > q) s0 = -3.0e38f;
        st0[r] = s0;
        cmax = fmaxf(cmax, s0);
        float s1 = (float)st1[r] * 0.125f;
        if (key0 + 16 > q) s1 = -3.0e38f;
        st1[r] = s1;
        cmax = fmaxf(cmax, s1);
      }
      cmax = fmaxf(cmax, __shfl_xor(cmax, 16, 32));
      const float mNew  = fmaxf(m, cmax);
      const float alpha = __expf(m - mNew);
      m = mNew;

      v16h ap;
      float rsum = 0.f;
#pragma unroll
      for (int r = 0; r < 8; ++r) {
        const float p0 = __expf((float)st0[r] - mNew);
        const float p1 = __expf((float)st1[r] - mNew);
        rsum += p0 + p1;
        ap[r]     = (_Float16)p0;
        ap[8 + r] = (_Float16)p1;
      }
      rsum += __shfl_xor(rsum, 16, 32);
      lsum = lsum * alpha + rsum;

      float av[8];
#pragma unroll
      for (int r = 0; r < 8; ++r) av[r] = __shfl(alpha, r + 8 * half, 32);
#pragma unroll
      for (int t = 0; t < 4; ++t)
#pragma unroll
        for (int r = 0; r < 8; ++r) o[t][r] *= av[r];

#pragma unroll
      for (int t = 0; t < 4; ++t) {
        const _Float16* vr = vTile + (size_t)(t * 16 + lq) * vstr + 8 * half;
        v16h bv = pack16(*(const v8h*)(vr), *(const v8h*)(vr + 16));
        o[t] = wmma_f16(ap, bv, o[t]);
      }
    }
  }

  // epilogue: divide by row sums, store f16 attn [B,S,D]
  float lv[8];
#pragma unroll
  for (int r = 0; r < 8; ++r) lv[r] = __shfl(lsum, r + 8 * half, 32);
  const int bIdx = bh >> 3, h = bh & 7;
#pragma unroll
  for (int t = 0; t < 4; ++t) {
    const int col = h * DHEAD + t * 16 + lq;
    _Float16* dst = attn + ((size_t)bIdx * S_LEN + qBase + 8 * half) * DMODEL + col;
#pragma unroll
    for (int r = 0; r < 8; ++r)
      dst[(size_t)r * DMODEL] = (_Float16)(o[t][r] / lv[r]);
  }
}

// ---------------------------------------------------------------------------
// Kernel 3: output projection  y = attn @ W_proj^T + b_proj (f32 out),
// 16x64 tile per wave.
// ---------------------------------------------------------------------------
__global__ void __launch_bounds__(256)
proj_kernel(const _Float16* __restrict__ attn,
            const _Float16* __restrict__ wp,
            const float* __restrict__ bias,
            float* __restrict__ out) {
  const int lane = threadIdx.x & 31;
  const int wid  = blockIdx.x * (blockDim.x >> 5) + (threadIdx.x >> 5);
  if (wid >= 512 * 8) return;
  const int rowTile = wid >> 3;
  const int cg      = wid & 7;
  const int half = lane >> 4;
  const int lq   = lane & 15;

  const _Float16* ar = attn + (size_t)(rowTile * 16 + lq) * DMODEL;
  const _Float16* wr = wp + (size_t)(cg * 64 + lq) * DMODEL;

  v8f c[4] = {};
#pragma unroll 4
  for (int it = 0; it < 16; ++it) {
    const int ka = it * 32 + 16 * half;
    v16h a = pack16(*(const v8h*)(ar + ka), *(const v8h*)(ar + ka + 8));
    const int kbb = it * 32 + 16 * half;
#pragma unroll
    for (int t = 0; t < 4; ++t) {
      const _Float16* w = wr + (size_t)t * 16 * DMODEL + kbb;
      v16h b = pack16(*(const v8h*)(w), *(const v8h*)(w + 8));
      c[t] = wmma_f16(a, b, c[t]);
    }
  }

  float* dstRow = out + (size_t)(rowTile * 16 + 8 * half) * DMODEL;
#pragma unroll
  for (int t = 0; t < 4; ++t) {
    const int e  = cg * 64 + t * 16 + lq;
    const float bv = bias[e];
#pragma unroll
    for (int r = 0; r < 8; ++r) dstRow[(size_t)r * DMODEL + e] = c[t][r] + bv;
  }
}

// ---------------------------------------------------------------------------
extern "C" void kernel_launch(void* const* d_in, const int* in_sizes, int n_in,
                              void* d_out, int out_size, void* d_ws, size_t ws_size,
                              hipStream_t stream) {
  const float* x   = (const float*)d_in[0];
  const float* Wq  = (const float*)d_in[1];
  const float* Wk  = (const float*)d_in[2];
  const float* Wv  = (const float*)d_in[3];
  const float* Wp  = (const float*)d_in[4];
  const float* bpj = (const float*)d_in[5];

  const size_t NX = (size_t)2 * S_LEN * DMODEL;  // 4,194,304
  const size_t NW = (size_t)DMODEL * DMODEL;     //   262,144
  _Float16* ws  = (_Float16*)d_ws;
  _Float16* xh  = ws;
  _Float16* wqh = xh + NX;
  _Float16* wkh = wqh + NW;
  _Float16* wvh = wkh + NW;
  _Float16* wph = wvh + NW;
  _Float16* qb  = wph + NW;
  _Float16* kb  = qb + NX;
  _Float16* vtb = kb + NX;
  _Float16* att = xh;   // alias: xh dead after qkv_kernel   (~34 MB total)

  cvt_kernel<<<(int)(NX / 8 + 255) / 256, 256, 0, stream>>>(x, xh, (int)(NX / 8));
  cvt_kernel<<<(int)(NW / 8 + 255) / 256, 256, 0, stream>>>(Wq, wqh, (int)(NW / 8));
  cvt_kernel<<<(int)(NW / 8 + 255) / 256, 256, 0, stream>>>(Wk, wkh, (int)(NW / 8));
  cvt_kernel<<<(int)(NW / 8 + 255) / 256, 256, 0, stream>>>(Wv, wvh, (int)(NW / 8));
  cvt_kernel<<<(int)(NW / 8 + 255) / 256, 256, 0, stream>>>(Wp, wph, (int)(NW / 8));

  qkv_kernel<<<1536, 256, 0, stream>>>(xh, wqh, wkh, wvh, qb, kb, vtb);
  flash_kernel<<<512, 256, 8192, stream>>>(qb, kb, vtb, att);
  proj_kernel<<<512, 256, 0, stream>>>(att, wph, bpj, (float*)d_out);
}
